// Sparse_Connect_1374389534850
// MI455X (gfx1250) — matrix operand order repaired and checked by reading, not compiled, not run on hardware
//
#include <hip/hip_runtime.h>

// ---------------------------------------------------------------------------
// DeepR sparse-connect forward on MI455X (gfx1250, wave32, WMMA + TDM).
// y = relu(relu(x @ W1) @ W2), W = th*sign*(th>0), D=1024, M=32768.
// Split-precision f16 (hi+lo) WMMA with f32 accumulation; all operands are
// pre-split to f16 pairs so the GEMM inner loop is ds_load_b128 + WMMA only.
// Staging: Tensor Data Mover (tensor_load_to_lds) double-buffered into LDS.
// ---------------------------------------------------------------------------

typedef __attribute__((ext_vector_type(16))) _Float16 v16h;
typedef __attribute__((ext_vector_type(4)))  _Float16 v4h;
typedef __attribute__((ext_vector_type(8)))  float    v8f;
typedef __attribute__((ext_vector_type(4)))  unsigned v4u;
typedef __attribute__((ext_vector_type(8)))  unsigned v8u;

#define D_DIM 1024
#define M_DIM 32768
#define TM 64       // workgroup M tile
#define TN 128      // workgroup N tile
#define TK 32       // K step (one f16 WMMA K depth)
#define HP 40       // LDS pitch for all f16 tiles, halves (80B = 16dw + 4dw TDM pad)

// --- split an f32 array into f16 hi/lo pair arrays (4 elems/thread)
__global__ __launch_bounds__(256) void split_f32_to_f16pair(
    const float* __restrict__ in, _Float16* __restrict__ hi, _Float16* __restrict__ lo)
{
    size_t i = (size_t)blockIdx.x * 256 + threadIdx.x;
    const float4 v = ((const float4*)in)[i];
    _Float16 h0 = (_Float16)v.x, h1 = (_Float16)v.y;
    _Float16 h2 = (_Float16)v.z, h3 = (_Float16)v.w;
    v4h H = {h0, h1, h2, h3};
    v4h L = {(_Float16)(v.x - (float)h0), (_Float16)(v.y - (float)h1),
             (_Float16)(v.z - (float)h2), (_Float16)(v.w - (float)h3)};
    ((v4h*)hi)[i] = H;
    ((v4h*)lo)[i] = L;
}

// --- prep: reconstruct w = th*sign*(th>0), split into f16 hi/lo, store [N][K]
__global__ __launch_bounds__(256) void deepr_prep_weights(
    const float* __restrict__ th, const float* __restrict__ sign,
    _Float16* __restrict__ whi, _Float16* __restrict__ wlo)
{
    int i = blockIdx.x * 256 + threadIdx.x;      // i over D*D, row-major [K][N]
    int k = i >> 10;
    int n = i & (D_DIM - 1);
    float t = th[i];
    float w = (t > 0.0f) ? t * sign[i] : 0.0f;
    _Float16 h = (_Float16)w;
    float lo = w - (float)h;
    // transposed store: [N][K]; scattered 2B writes land in a 2MB region -> L2 absorbs
    whi[(size_t)n * D_DIM + k] = h;
    wlo[(size_t)n * D_DIM + k] = (_Float16)lo;
}

// --- TDM: issue one 2-D tile load (global -> LDS) via descriptor in SGPRs.
//     pad_interval code pi pads after 2^(pi+1) dwords; pad_amount code pa pads pa+1 dwords.
__device__ __forceinline__ void tdm_load_tile_f16(
    unsigned lds_addr, const void* gptr,
    unsigned tile_d1)                          // #rows; rows are TK halves, stride D_DIM
{
    unsigned long long ga = (unsigned long long)(uintptr_t)gptr;
    v4u g0;
    g0.x = 1u;                                            // count=1 (user D#)
    g0.y = lds_addr;                                      // [63:32] lds_addr
    g0.z = (unsigned)(ga & 0xFFFFFFFFu);                  // global_addr lo
    g0.w = ((unsigned)(ga >> 32) & 0x01FFFFFFu)           // global_addr[56:32]
         | (2u << 30);                                    // type = 2 ("image")
    v8u g1;
    g1.s0 = (1u << 16)                                    // data_size = 2B
          | (1u << 20)                                    // pad_enable
          | (3u << 22)                                    // pad_interval: 16 dwords
          | (3u << 25);                                   // pad_amount: 4 dwords -> 80B pitch
    g1.s1 = ((unsigned)D_DIM & 0xFFFFu) << 16;            // tensor_dim0[15:0]
    g1.s2 = (((unsigned)D_DIM >> 16) & 0xFFFFu)
          | (((unsigned)M_DIM & 0xFFFFu) << 16);          // tensor_dim1[15:0]
    g1.s3 = (((unsigned)M_DIM >> 16) & 0xFFFFu)
          | (((unsigned)TK & 0xFFFFu) << 16);             // tile_dim0 = 32 halves
    g1.s4 = (tile_d1 & 0xFFFFu);                          // tile_dim1; tile_dim2=0
    g1.s5 = (unsigned)D_DIM;                              // tensor_dim0_stride lo32
    g1.s6 = 0u;
    g1.s7 = 0u;
    asm volatile("tensor_load_to_lds %0, %1" :: "s"(g0), "s"(g1) : "memory");
}

__device__ __forceinline__ unsigned lds_off(const void* p) {
    return (unsigned)(uintptr_t)p;    // low 32 bits of flat addr = LDS byte offset
}

// assemble a v16h fragment from two 16B LDS vector loads (forces 2x ds_load_b128)
__device__ __forceinline__ v16h frag_from_lds(const unsigned* base32, int di0, int di1) {
    union { v16h h; v4u q[2]; } f;
    f.q[0] = *(const v4u*)(base32 + di0);
    f.q[1] = *(const v4u*)(base32 + di1);
    return f.h;
}

// --- one layer: Y = act(A @ W^T-stored) ; A given as split-f16 [M][K] pair,
//     W given transposed+split [N][K] pair. SPLIT_OUT: write split-f16 Y
//     (feeds next layer), else write f32 Y (final output).
template<bool SPLIT_OUT>
__global__ __launch_bounds__(256) void deepr_gemm_layer(
    const _Float16* __restrict__ Ahi, const _Float16* __restrict__ Alo,
    const _Float16* __restrict__ Whi, const _Float16* __restrict__ Wlo,
    float* __restrict__ Yf, _Float16* __restrict__ Yhi, _Float16* __restrict__ Ylo,
    const int* __restrict__ reluFlag)
{
    __shared__ __align__(16) _Float16 AsH[2 * TM * HP];
    __shared__ __align__(16) _Float16 AsL[2 * TM * HP];
    __shared__ __align__(16) _Float16 WsH[2 * TN * HP];
    __shared__ __align__(16) _Float16 WsL[2 * TN * HP];

    const int t      = threadIdx.x;
    const int m0     = blockIdx.y * TM;
    const int n0     = blockIdx.x * TN;
    const int wave   = t >> 5;
    const int lane   = t & 31;
    const int l16    = lane & 15;
    const int laneHi = (lane >> 4) & 1;   // 0: lanes 0-15, 1: lanes 16-31
    const int wm     = wave & 1;          // wave M slot (0..1)  -> 32 rows
    const int wn     = wave >> 1;         // wave N slot (0..3)  -> 32 cols

    v8f acc[2][2];
    #pragma unroll
    for (int i = 0; i < 2; ++i)
        #pragma unroll
        for (int j = 0; j < 2; ++j)
            acc[i][j] = (v8f){0.f, 0.f, 0.f, 0.f, 0.f, 0.f, 0.f, 0.f};

    // Issue all 4 tile DMAs for one K-step into buffer `buf` (wave 0 only).
    auto issue_tiles = [&](int buf, int kb) {
        tdm_load_tile_f16(lds_off(&AsH[buf * TM * HP]), Ahi + (size_t)m0 * D_DIM + kb, TM);
        tdm_load_tile_f16(lds_off(&AsL[buf * TM * HP]), Alo + (size_t)m0 * D_DIM + kb, TM);
        tdm_load_tile_f16(lds_off(&WsH[buf * TN * HP]), Whi + (size_t)n0 * D_DIM + kb, TN);
        tdm_load_tile_f16(lds_off(&WsL[buf * TN * HP]), Wlo + (size_t)n0 * D_DIM + kb, TN);
    };

    if (wave == 0) issue_tiles(0, 0);                    // prologue fill buf 0

    for (int kb = 0, it = 0; kb < D_DIM; kb += TK, ++it) {
        const int  buf     = it & 1;
        const bool hasNext = (kb + TK) < D_DIM;

        if (wave == 0 && hasNext) issue_tiles(buf ^ 1, kb + TK);
        // TDM completes in-order per wave: <=4 outstanding => current buf done.
        if (hasNext) __builtin_amdgcn_s_wait_tensorcnt(4);
        else         __builtin_amdgcn_s_wait_tensorcnt(0);
        __syncthreads();                                 // buf visible to all waves

        const unsigned* AH32 = (const unsigned*)(&AsH[buf * TM * HP]);
        const unsigned* AL32 = (const unsigned*)(&AsL[buf * TM * HP]);
        const unsigned* WH32 = (const unsigned*)(&WsH[buf * TN * HP]);
        const unsigned* WL32 = (const unsigned*)(&WsL[buf * TN * HP]);

        // ---- A fragments (16x32 f16: lanes 0-15 M; VGPR r -> K pair; K+8 hi lanes)
        //      dwords [base .. base+3] then [base+8 .. base+11]
        v16h ahi[2], alo[2];
        #pragma unroll
        for (int mt = 0; mt < 2; ++mt) {
            int m    = wm * 32 + mt * 16 + l16;
            int base = m * (HP / 2) + laneHi * 4;        // dword index
            ahi[mt] = frag_from_lds(AH32, base, base + 8);
            alo[mt] = frag_from_lds(AL32, base, base + 8);
        }
        // ---- B fragments (32x16 f16: same K order as A; K+8 hi lanes)
        //      dwords [base .. base+3] then [base+8 .. base+11]
        v16h bhi[2], blo[2];
        #pragma unroll
        for (int nt = 0; nt < 2; ++nt) {
            int n    = wn * 32 + nt * 16 + l16;
            int base = n * (HP / 2) + laneHi * 4;        // dword index
            bhi[nt] = frag_from_lds(WH32, base, base + 8);
            blo[nt] = frag_from_lds(WL32, base, base + 8);
        }
        __syncthreads();   // all waves done reading buf; safe to re-fill next iter

        // ---- 12 WMMAs: hi*hi + hi*lo + lo*hi  (split-precision accumulate)
        #pragma unroll
        for (int mt = 0; mt < 2; ++mt) {
            #pragma unroll
            for (int nt = 0; nt < 2; ++nt) {
                acc[mt][nt] = __builtin_amdgcn_wmma_f32_16x16x32_f16(
                    false, ahi[mt], false, bhi[nt], (short)0, acc[mt][nt], false, false);
                acc[mt][nt] = __builtin_amdgcn_wmma_f32_16x16x32_f16(
                    false, ahi[mt], false, blo[nt], (short)0, acc[mt][nt], false, false);
                acc[mt][nt] = __builtin_amdgcn_wmma_f32_16x16x32_f16(
                    false, alo[mt], false, bhi[nt], (short)0, acc[mt][nt], false, false);
            }
        }
    }

    // ---- epilogue: optional relu; store f32 (final) or split-f16 (next layer)
    const bool doRelu = (reluFlag[0] != 0);
    #pragma unroll
    for (int mt = 0; mt < 2; ++mt) {
        #pragma unroll
        for (int nt = 0; nt < 2; ++nt) {
            int col = n0 + wn * 32 + nt * 16 + l16;
            #pragma unroll
            for (int j = 0; j < 8; ++j) {
                int row = m0 + wm * 32 + mt * 16 + j + (laneHi ? 8 : 0);
                float v = acc[mt][nt][j];
                if (doRelu) v = (v > 0.0f) ? v : 0.0f;
                size_t idx = (size_t)row * D_DIM + col;
                if (SPLIT_OUT) {
                    _Float16 h = (_Float16)v;
                    Yhi[idx] = h;
                    Ylo[idx] = (_Float16)(v - (float)h);
                } else {
                    Yf[idx] = v;
                }
            }
        }
    }
}

extern "C" void kernel_launch(void* const* d_in, const int* in_sizes, int n_in,
                              void* d_out, int out_size, void* d_ws, size_t ws_size,
                              hipStream_t stream) {
    const float* x     = (const float*)d_in[0];
    const float* th1   = (const float*)d_in[1];
    const float* sign1 = (const float*)d_in[2];
    const float* th2   = (const float*)d_in[3];
    const float* sign2 = (const float*)d_in[4];
    const int*   relu  = (const int*)  d_in[5];

    const size_t MD = (size_t)M_DIM * D_DIM;

    // d_out (128MB f32) doubles as scratch for the x split (consumed by layer 1,
    // then overwritten by layer 2's final f32 output).
    _Float16* xhi  = (_Float16*)d_out;           // 64MB
    _Float16* xlo  = xhi + MD;                   // 64MB
    float*    outf = (float*)d_out;

    // d_ws: [ y1hi 64MB ][ y1lo 64MB ][ W1hi W1lo W2hi W2lo : 2MB each ]
    _Float16* y1hi = (_Float16*)d_ws;
    _Float16* y1lo = y1hi + MD;
    _Float16* w1hi = y1lo + MD;
    _Float16* w1lo = w1hi + (size_t)D_DIM * D_DIM;
    _Float16* w2hi = w1lo + (size_t)D_DIM * D_DIM;
    _Float16* w2lo = w2hi + (size_t)D_DIM * D_DIM;

    split_f32_to_f16pair<<<(int)(MD / 4 / 256), 256, 0, stream>>>(x, xhi, xlo);
    const int prepBlocks = (D_DIM * D_DIM) / 256;
    deepr_prep_weights<<<prepBlocks, 256, 0, stream>>>(th1, sign1, w1hi, w1lo);
    deepr_prep_weights<<<prepBlocks, 256, 0, stream>>>(th2, sign2, w2hi, w2lo);

    dim3 grid(D_DIM / TN, M_DIM / TM);   // (8, 512)
    deepr_gemm_layer<true ><<<grid, 256, 0, stream>>>(xhi,  xlo,  w1hi, w1lo,
                                                      nullptr, y1hi, y1lo, relu);
    deepr_gemm_layer<false><<<grid, 256, 0, stream>>>(y1hi, y1lo, w2hi, w2lo,
                                                      outf, nullptr, nullptr, relu);
}
